// BilateralSliceApply_78348793414239
// MI455X (gfx1250) — matrix operand order repaired and checked by reading, not compiled, not run on hardware
//
#include <hip/hip_runtime.h>

typedef __attribute__((ext_vector_type(16))) _Float16 v16h;
typedef __attribute__((ext_vector_type(8)))  _Float16 v8h;
typedef __attribute__((ext_vector_type(4)))  _Float16 v4h;
typedef __attribute__((ext_vector_type(8)))  float    v8f;

#define BB    2
#define COUT  8
#define CIN   8
#define DD    8
#define GHH   16
#define GWW   16
#define HH    512
#define WW    512
#define HW    (HH*WW)
#define KTOT  64

// out[b,c,h,w] = sum_ci inp*weight + mean_ci bias, weight/bias = trilinear slice of grid.
// Factorized: out[c,pix] = G[c, (k,d,j)] . F[(k,d,j), pix], G per grid-cell, F per pixel.
__global__ __launch_bounds__(256, 2)
void bilateral_slice_wmma(const float* __restrict__ grid,
                          const float* __restrict__ guide,
                          const float* __restrict__ inp,
                          float* __restrict__ out)
{
    const int cx = blockIdx.x;   // 0..14 (grid cell x)
    const int cy = blockIdx.y;   // 0..14 (grid cell y)
    const int b  = blockIdx.z;   // batch

    __shared__ _Float16 Ald[16][KTOT];    // A matrix (rows 8..15 = 0)
    __shared__ _Float16 Fld[256][KTOT];   // per-pixel feature columns

    const int t    = threadIdx.x;
    const int lane = t & 31;
    const int half = lane >> 4;
    const int ln16 = lane & 15;
    const int wave = t >> 5;

    // ---- stage per-cell G matrix (grid corner values) into LDS ----
    #pragma unroll
    for (int i = 0; i < 4; ++i) {
        int e = t + i * 256;            // 16*64 = 1024 entries
        int m = e >> 6;
        int K = e & 63;                 // K = kk*32 + d*4 + j
        _Float16 v = (_Float16)0.0f;
        if (m < COUT) {
            int kk = K >> 5;
            int d  = (K >> 2) & 7;
            int j  = K & 3;
            int gh = cy + ((j >> 1) & 1);
            int gw = cx + (j & 1);
            int gi = ((((b*COUT + m)*2 + kk)*DD + d)*GHH + gh)*GWW + gw;
            v = (_Float16)grid[gi];
        }
        Ald[m][K] = v;
    }
    __syncthreads();

    // pixel region owned by this cell: floor(h*15/511) == cy (with edge clip)
    const int h0 = (511*cy + 14) / 15;
    const int h1 = (cy == 14) ? 512 : (511*(cy+1) + 14) / 15;
    const int w0 = (511*cx + 14) / 15;
    const int w1 = (cx == 14) ? 512 : (511*(cx+1) + 14) / 15;
    const int RW = w1 - w0;
    const int Np = (h1 - h0) * RW;
    const int nG = (Np + 15) >> 4;      // 16-pixel tiles

    for (int g0 = wave * 2; g0 < nG; g0 += 16) {
        const int g1 = g0 + 1;

        // ---- per-lane pixel features (lanes 0-15 -> tile g0, 16-31 -> tile g1) ----
        const int gsel = (lane < 16) ? g0 : g1;
        int p = gsel * 16 + ln16;
        const bool pvalid = (gsel < nG) && (p < Np);
        int pc = pvalid ? p : 0;
        int py = pc / RW;
        int px = pc - py * RW;
        int h = h0 + py, w = w0 + px;

        float yp = (float)h * (15.0f / 511.0f);
        float xp = (float)w * (15.0f / 511.0f);
        float wy = yp - (float)cy;
        float wx = xp - (float)cx;
        float w00 = (1.f - wy) * (1.f - wx), w01 = (1.f - wy) * wx;
        float w10 = wy * (1.f - wx),         w11 = wy * wx;

        float Sw[8], Sb[8];
        #pragma unroll
        for (int d = 0; d < 8; ++d) { Sw[d] = 0.f; Sb[d] = 0.f; }
        const int base = b * CIN * HW + h * WW + w;
        #pragma unroll
        for (int ci = 0; ci < CIN; ++ci) {
            float gv = guide[base + ci * HW];
            float xv = inp  [base + ci * HW];
            float zp = fminf(fmaxf(gv * 7.0f, 0.0f), 7.0f);
            #pragma unroll
            for (int d = 0; d < 8; ++d) {
                float tt = fmaxf(0.0f, 1.0f - fabsf(zp - (float)d));
                Sw[d] = fmaf(xv, tt, Sw[d]);
                Sb[d] += tt;
            }
        }
        _Float16* frow = &Fld[t][0];
        #pragma unroll
        for (int d = 0; d < 8; ++d) {
            float s0 = Sw[d];
            *(v4h*)&frow[d * 4] = (v4h){(_Float16)(s0*w00), (_Float16)(s0*w01),
                                        (_Float16)(s0*w10), (_Float16)(s0*w11)};
            float s1 = Sb[d] * 0.125f;
            *(v4h*)&frow[32 + d * 4] = (v4h){(_Float16)(s1*w00), (_Float16)(s1*w01),
                                             (_Float16)(s1*w10), (_Float16)(s1*w11)};
        }
        // per-wave LDS RAW fence (DS ops are in-order per wave; keep compiler honest)
        asm volatile("s_wait_dscnt 0" ::: "memory");

        // ---- WMMA: D(16x16,f32) = A(16x64,f16) x F(64x16,f16), two pixel tiles ----
        v8f acc0 = {}; v8f acc1 = {};
        const int rowb = t & ~31;       // this wave's Fld slot base
        #pragma unroll
        for (int kb = 0; kb < 2; ++kb) {
            const int kbase = kb * 32;
            // A 16x32 f16 layout: lane(m,half): elems 0-15 = K kbase+half*16.. for row m
            v8h alo = *(const v8h*)&Ald[ln16][kbase + half * 16];
            v8h ahi = *(const v8h*)&Ald[ln16][kbase + half * 16 + 8];
            v16h a = __builtin_shufflevector(alo, ahi,
                     0,1,2,3,4,5,6,7,8,9,10,11,12,13,14,15);
            // B 32x16 f16 layout: lane(n,half): elems 0-15 = K kbase+half*16.. for column n
            v16h bf0 = *(const v16h*)&Fld[rowb + ln16][kbase + half * 16];
            v16h bf1 = *(const v16h*)&Fld[rowb + 16 + ln16][kbase + half * 16];
            acc0 = __builtin_amdgcn_wmma_f32_16x16x32_f16(false, a, false, bf0,
                                                          (short)0, acc0, false, false);
            acc1 = __builtin_amdgcn_wmma_f32_16x16x32_f16(false, a, false, bf1,
                                                          (short)0, acc1, false, false);
        }

        // ---- store: lanes 0-15 hold c_out 0..7 (elems 0..7) for their pixel column ----
        if (lane < 16) {
            int p0 = g0 * 16 + lane;
            if (p0 < Np) {
                int qy = p0 / RW, qx = p0 - qy * RW;
                int oh = h0 + qy, ow = w0 + qx;
                int ob = b * COUT * HW + oh * WW + ow;
                #pragma unroll
                for (int r = 0; r < 8; ++r) out[ob + r * HW] = acc0[r];
            }
            int p1 = g1 * 16 + lane;
            if (g1 < nG && p1 < Np) {
                int qy = p1 / RW, qx = p1 - qy * RW;
                int oh = h0 + qy, ow = w0 + qx;
                int ob = b * COUT * HW + oh * WW + ow;
                #pragma unroll
                for (int r = 0; r < 8; ++r) out[ob + r * HW] = acc1[r];
            }
        }
    }
}

extern "C" void kernel_launch(void* const* d_in, const int* in_sizes, int n_in,
                              void* d_out, int out_size, void* d_ws, size_t ws_size,
                              hipStream_t stream) {
    const float* grid  = (const float*)d_in[0];
    const float* guide = (const float*)d_in[1];
    const float* inp   = (const float*)d_in[2];
    float* out = (float*)d_out;
    dim3 g(15, 15, BB);   // one block per (cell_x, cell_y, batch)
    bilateral_slice_wmma<<<g, 256, 0, stream>>>(grid, guide, inp, out);
}
